// SOMFNN_59571196395768
// MI455X (gfx1250) — matrix operand order repaired and checked by reading, not compiled, not run on hardware
//
#include <hip/hip_runtime.h>
#include <stdint.h>

typedef _Float16 h16v __attribute__((ext_vector_type(16)));
typedef _Float16 h8v  __attribute__((ext_vector_type(8)));
typedef _Float16 h4v  __attribute__((ext_vector_type(4)));
typedef float    f8v  __attribute__((ext_vector_type(8)));

#define S_TOT 8192
#define DIN   1024
#define NR    64
#define NO    64
#define MT    64                     // sample rows per workgroup
#define KC    512                    // K-chunk of W staged in LDS per step
#define PAD   8                      // halfs of padding per LDS row (bank spread)
#define XROW  (DIN + PAD)            // 1032 halfs -> stride 2064 B (516 dw, %64 = 4)
#define WROW  (KC + PAD)             // 520 halfs  -> stride 1040 B (260 dw, %64 = 4)
#define XS_BYTES (MT * XROW * 2)     // 132096 B
#define LS_BYTES (MT * NR * 4)       // 16384 B
#define WBUF_BYTES (64 * WROW * 2)   // 66560 B per buffer
#define WS_OFF   (XS_BYTES + LS_BYTES)
#define SMEM_BYTES (WS_OFF + 2 * WBUF_BYTES)   // 281600 B <= 320 KB WGP LDS

__device__ __forceinline__ float fast_sigmoid(float x) {
    const float e = __builtin_amdgcn_exp2f(x * -1.44269504088896340736f);
    return __builtin_amdgcn_rcpf(1.0f + e);
}

// ---- prep: convert W fp32 [4096,1024] -> f16 row-major in workspace ----
__global__ __launch_bounds__(256) void somfnn_w2h(const float4* __restrict__ w4,
                                                  h4v* __restrict__ wh4, int n4) {
    int i = blockIdx.x * blockDim.x + threadIdx.x;
    if (i < n4) {
        float4 v = w4[i];
        h4v h = { (_Float16)v.x, (_Float16)v.y, (_Float16)v.z, (_Float16)v.w };
        wh4[i] = h;
    }
}

// Issue async global->LDS copies for one W chunk: rows r*64..r*64+63, cols kc..kc+KC.
// Each of the 256 threads copies 16 x 16B units; per wave that is 16 async b128 ops.
__device__ __forceinline__ void stage_w_async(const _Float16* __restrict__ Wh,
                                              uint32_t lds_off, int r, int kc, int tid) {
    const int ku  = tid & 63;            // 16B unit within a row (64 units = 512 halfs)
    const int row = tid >> 6;            // starting row 0..3, advances by 4
    const _Float16* sbase = Wh + (size_t)r * 64 * DIN + kc;
    uint32_t goff = (uint32_t)row * (DIN * 2) + (uint32_t)ku * 16;
    uint32_t loff = lds_off + (uint32_t)row * (WROW * 2) + (uint32_t)ku * 16;
    #pragma unroll
    for (int it = 0; it < 16; ++it) {
        asm volatile("global_load_async_to_lds_b128 %0, %1, %2"
                     :: "v"(loff), "v"(goff), "s"(sbase) : "memory");
        goff += 4 * DIN * 2;
        loff += 4 * WROW * 2;
    }
}

// ---- fused: Y[s,o] = sum_r lambda[s,r] * sigmoid(X W_r^T + b_r) ----
__global__ __launch_bounds__(256) void somfnn_gemm(const float* __restrict__ X,
                                                   const _Float16* __restrict__ Wh,
                                                   const float* __restrict__ bias,
                                                   const float* __restrict__ lam,
                                                   float* __restrict__ Y) {
    extern __shared__ char smem[];
    _Float16* Xs = (_Float16*)smem;              // [MT][XROW] f16
    float*    Ls = (float*)(smem + XS_BYTES);    // [MT][NR]   f32
    const uint32_t smem_base = (uint32_t)(uintptr_t)(void*)smem;  // LDS byte offset

    const int tid = threadIdx.x;
    const int m0  = blockIdx.x * MT;

    // Stage X tile, converting fp32 -> f16 (each element read exactly once).
    for (int e = tid * 4; e < MT * DIN; e += 256 * 4) {
        const int row = e >> 10, col = e & (DIN - 1);
        const float4 v = *(const float4*)(X + (size_t)(m0 + row) * DIN + col);
        h4v h = { (_Float16)v.x, (_Float16)v.y, (_Float16)v.z, (_Float16)v.w };
        *(h4v*)(Xs + row * XROW + col) = h;
    }
    // Stage lambda tile.
    for (int e = tid; e < MT * NR; e += 256) {
        Ls[e] = lam[(size_t)(m0 + (e >> 6)) * NR + (e & 63)];
    }
    __syncthreads();

    const int lane   = tid & 31;
    const int wid    = tid >> 5;
    const int wm     = wid & 3;    // 16-row block within tile
    const int wn     = wid >> 2;   // 32-col block within the 64 outputs
    const int lane_m = lane & 15;
    const int hi     = lane >> 4;

    // A fragment source: row (wm*16 + lane_m) of the LDS X tile.
    const _Float16* arow = Xs + (wm * 16 + lane_m) * XROW;

    f8v y0 = {}; f8v y1 = {};
    f8v acc0 = {}; f8v acc1 = {};

    // Prologue: stage chunk g=0 (rule 0, kc 0) into buffer 0.
    stage_w_async(Wh, smem_base + WS_OFF, 0, 0, tid);

    const int NG = NR * (DIN / KC);    // 128 chunks total
    for (int g = 0; g < NG; ++g) {
        // Overlap: issue copies for chunk g+1 into the other buffer, then wait
        // until only those 16 remain outstanding (=> chunk g complete, in-order).
        if (g + 1 < NG) {
            const int gn = g + 1;
            stage_w_async(Wh, smem_base + WS_OFF + (uint32_t)(gn & 1) * WBUF_BYTES,
                          gn >> 1, (gn & 1) * KC, tid);
            asm volatile("s_wait_asynccnt 0x10" ::: "memory");
        } else {
            asm volatile("s_wait_asynccnt 0x0" ::: "memory");
        }
        __syncthreads();   // all waves' chunk-g data visible in LDS

        const int cc = g & 1;          // K-chunk within rule
        const int r  = g >> 1;
        if (cc == 0) { acc0 = (f8v){}; acc1 = (f8v){}; }

        const _Float16* wb = (const _Float16*)(smem + WS_OFF + (size_t)cc * WBUF_BYTES);
        const _Float16* brow0 = wb + (wn * 32 + lane_m) * WROW + hi * 16;
        const _Float16* brow1 = brow0 + 16 * WROW;
        const _Float16* arowc = arow + cc * KC;

        #pragma unroll 4
        for (int ks = 0; ks < KC / 32; ++ks) {
            const int k0 = ks * 32;
            union { h16v v; h8v h[2]; } a, b0, b1;
            // A (16x32 f16): lane holds K {k0+hi*16 .. k0+hi*16+15} contiguous
            a.h[0]  = *(const h8v*)(arowc + k0 + hi * 16);
            a.h[1]  = *(const h8v*)(arowc + k0 + hi * 16 + 8);
            // B (32x16 f16): lane holds K {k0+hi*16 .. k0+hi*16+15} contiguous
            b0.h[0] = *(const h8v*)(brow0 + k0);
            b0.h[1] = *(const h8v*)(brow0 + k0 + 8);
            b1.h[0] = *(const h8v*)(brow1 + k0);
            b1.h[1] = *(const h8v*)(brow1 + k0 + 8);
            acc0 = __builtin_amdgcn_wmma_f32_16x16x32_f16(false, a.v, false, b0.v,
                                                          (short)0, acc0, false, false);
            acc1 = __builtin_amdgcn_wmma_f32_16x16x32_f16(false, a.v, false, b1.v,
                                                          (short)0, acc1, false, false);
        }

        if (cc == 1) {
            // Epilogue for rule r: bias + sigmoid + lambda-weighted accumulate.
            const int colbase = wn * 32;
            const float bias0 = bias[r * 64 + colbase + lane_m];
            const float bias1 = bias[r * 64 + colbase + 16 + lane_m];
            #pragma unroll
            for (int j = 0; j < 8; ++j) {
                // C/D layout: VGPR j, lane -> (row = j + 8*hi, col = lane&15)
                const float lv = Ls[(wm * 16 + hi * 8 + j) * NR + r];
                y0[j] += lv * fast_sigmoid(acc0[j] + bias0);
                y1[j] += lv * fast_sigmoid(acc1[j] + bias1);
            }
        }
        __syncthreads();   // all waves done reading buf[cc] before it is re-staged
    }

    #pragma unroll
    for (int j = 0; j < 8; ++j) {
        const int row = m0 + wm * 16 + hi * 8 + j;
        float* yp = Y + (size_t)row * NO + wn * 32 + lane_m;
        yp[0]  = y0[j];
        yp[16] = y1[j];
    }
}

extern "C" void kernel_launch(void* const* d_in, const int* in_sizes, int n_in,
                              void* d_out, int out_size, void* d_ws, size_t ws_size,
                              hipStream_t stream) {
    const float* X   = (const float*)d_in[0];
    const float* W   = (const float*)d_in[1];
    const float* b   = (const float*)d_in[2];
    const float* lam = (const float*)d_in[3];
    float* Y = (float*)d_out;
    _Float16* Wh = (_Float16*)d_ws;   // needs 4096*1024*2 = 8 MB of workspace

    const int n4 = (NR * NO * DIN) / 4;
    somfnn_w2h<<<(n4 + 255) / 256, 256, 0, stream>>>((const float4*)W, (h4v*)Wh, n4);
    somfnn_gemm<<<S_TOT / MT, 256, SMEM_BYTES, stream>>>(X, Wh, b, lam, Y);
}
